// DescripNet_41351945126185
// MI455X (gfx1250) — matrix operand order repaired and checked by reading, not compiled, not run on hardware
//
#include <hip/hip_runtime.h>
#include <hip/hip_bf16.h>

#define BB   16
#define NN   2048
#define KK   16
#define BN   (BB * NN)
#define OUTC 256
#define BN_EPS 1e-5f

typedef __attribute__((ext_vector_type(16))) __bf16 v16bf;
typedef __attribute__((ext_vector_type(8)))  float  v8f;

static __device__ __forceinline__ v8f wmma_bf16(v16bf a, v16bf b, v8f c) {
    return __builtin_amdgcn_wmma_f32_16x16x32_bf16(false, a, false, b, (short)0, c, false, false);
}

// A fragment (16x32 MxK) from row-major bf16 rows m0.., K chunk k0.
// Lane l (row = l&15, half = l>>4) holds K = {k0+h*8 .. +7, k0+16+h*8 .. +7}.
static __device__ __forceinline__ v16bf load_a(const __bf16* base, int ld, int m0, int k0, int lane) {
    const int r = lane & 15, h = lane >> 4;
    const __bf16* row = base + (size_t)(m0 + r) * ld + k0 + h * 8;
    v16bf a;
#pragma unroll
    for (int i = 0; i < 8; ++i) a[i] = row[i];
#pragma unroll
    for (int i = 0; i < 8; ++i) a[8 + i] = row[16 + i];
    return a;
}

// B fragment for X*Y^T: B[k][n] = Y[n][k]. Lane l (col = l&15, half = l>>4)
// holds K = {k0+h*8 .. +7, k0+16+h*8 .. +7} of Y row n0+c.
static __device__ __forceinline__ v16bf load_bT_bf16(const __bf16* base, int ld, int n0, int k0, int lane) {
    const int c = lane & 15, h = lane >> 4;
    const __bf16* row = base + (size_t)(n0 + c) * ld + k0 + h * 8;
    v16bf b;
#pragma unroll
    for (int i = 0; i < 8; ++i) b[i] = row[i];
#pragma unroll
    for (int i = 0; i < 8; ++i) b[8 + i] = row[16 + i];
    return b;
}

// ---- convert f32 weights [Cout x Cin] -> zero-padded bf16 [Cout x Cpad] (opt W1-W2) ----
__global__ __launch_bounds__(256) void convw_kernel(const float* __restrict__ W1, const float* __restrict__ W2,
                                                    int Cin, int Cpad, int total, __bf16* __restrict__ out) {
    int i = blockIdx.x * 256 + threadIdx.x;
    if (i >= total) return;
    int o = i / Cpad, k = i - o * Cpad;
    float v = 0.f;
    if (k < Cin) { v = W1[(size_t)o * Cin + k]; if (W2) v -= W2[(size_t)o * Cin + k]; }
    out[i] = (__bf16)v;
}

// ---- convert h -> zero-padded bf16 + squared norms ----
__global__ __launch_bounds__(256) void prep_kernel(const float* __restrict__ src, int srcStride,
                                                   int Cin, int Cpad,
                                                   __bf16* __restrict__ hbf, float* __restrict__ sq) {
    int node = blockIdx.x * 256 + threadIdx.x;
    if (node >= BN) return;
    const float* r = src + (size_t)node * srcStride;
    float s = 0.f;
    for (int c = 0; c < Cpad; ++c) {
        float v = (c < Cin) ? r[c] : 0.f;
        s += v * v;
        hbf[(size_t)node * Cpad + c] = (__bf16)v;
    }
    sq[node] = s;
}

// ---- fused pairwise-distance (WMMA Gram) + top-k. One wave per 16-row tile. ----
template <int CPAD>
__global__ __launch_bounds__(32) void knn_kernel(const __bf16* __restrict__ hbf,
                                                 const float* __restrict__ sq,
                                                 int* __restrict__ idx) {
    constexpr int NCH = CPAD / 32;
    __shared__ float dtile[16][17];
    __shared__ float bestd[16][KK];
    __shared__ int   besti[16][KK];
    const int b = blockIdx.y, tile = blockIdx.x, lane = threadIdx.x;
    const int m0 = tile * 16;
    const __bf16* hb = hbf + (size_t)b * NN * CPAD;
    const float*  sqb = sq + b * NN;

    v16bf afrag[NCH];
#pragma unroll
    for (int ch = 0; ch < NCH; ++ch) afrag[ch] = load_a(hb, CPAD, m0, ch * 32, lane);
    float sqm[8];
#pragma unroll
    for (int v = 0; v < 8; ++v) sqm[v] = sqb[m0 + v + (lane >> 4) * 8];

    if (lane < 16)
        for (int j = 0; j < KK; ++j) { bestd[lane][j] = 1e30f; besti[lane][j] = 0; }
    __syncthreads();

    for (int ct = 0; ct < NN / 16; ++ct) {
        if (ct + 1 < NN / 16)
            __builtin_prefetch(hb + (size_t)(ct * 16 + 16 + (lane & 15)) * CPAD, 0, 1);
        v8f acc = {};
#pragma unroll
        for (int ch = 0; ch < NCH; ++ch) {
            v16bf bfrag = load_bT_bf16(hb, CPAD, ct * 16, ch * 32, lane);
            acc = wmma_bf16(afrag[ch], bfrag, acc);
        }
        float sqn = sqb[ct * 16 + (lane & 15)];
#pragma unroll
        for (int v = 0; v < 8; ++v) {
            int m = v + (lane >> 4) * 8;
            dtile[m][lane & 15] = sqm[v] + sqn - 2.f * acc[v];
        }
        __syncthreads();
        if (lane < 16) {
            float worst = bestd[lane][KK - 1];
#pragma unroll 1
            for (int j = 0; j < 16; ++j) {
                float d = dtile[lane][j];
                if (d < worst) {
                    int p = KK - 1;
                    while (p > 0 && bestd[lane][p - 1] > d) {
                        bestd[lane][p] = bestd[lane][p - 1];
                        besti[lane][p] = besti[lane][p - 1];
                        --p;
                    }
                    bestd[lane][p] = d;
                    besti[lane][p] = ct * 16 + j;
                    worst = bestd[lane][KK - 1];
                }
            }
        }
        __syncthreads();
    }
    if (lane < 16)
        for (int j = 0; j < KK; ++j)
            idx[(size_t)(b * NN + m0 + lane) * KK + j] = besti[lane][j];
}

// ---- generic node GEMM: out = [relu]( h * Wbf^T + b1[+b2] ), out stride OUTC ----
template <int CPAD>
__global__ __launch_bounds__(256) void node_gemm_kernel(const __bf16* __restrict__ hbf,
                                                        const __bf16* __restrict__ Wbf,
                                                        const float* __restrict__ b1, const float* __restrict__ b2,
                                                        float* __restrict__ out, int Cout, int do_relu) {
    const int lane = threadIdx.x & 31;
    const int wid = blockIdx.x * 8 + (threadIdx.x >> 5);
    const int ctiles = Cout >> 4;
    const int nodeTile = wid / ctiles, coutTile = wid % ctiles;
    const int m0 = nodeTile * 16, n0 = coutTile * 16;
    v8f acc = {};
#pragma unroll
    for (int k0 = 0; k0 < CPAD; k0 += 32) {
        v16bf a = load_a(hbf, CPAD, m0, k0, lane);
        v16bf bfrag = load_bT_bf16(Wbf, CPAD, n0, k0, lane);
        acc = wmma_bf16(a, bfrag, acc);
    }
    const int n = n0 + (lane & 15);
    float bias = b1[n] + (b2 ? b2[n] : 0.f);
#pragma unroll
    for (int v = 0; v < 8; ++v) {
        int m = m0 + v + (lane >> 4) * 8;
        float y = acc[v] + bias;
        if (do_relu) y = fmaxf(y, 0.f);
        out[(size_t)m * OUTC + n] = y;
    }
}

// ---- per-node 16(edges) x 16(cout) tile: m = h_j*Wt^T + p(node) broadcast ----
template <int CPAD>
static __device__ __forceinline__ v8f edge_tile(const __bf16* __restrict__ hbf,
                                                const __bf16* __restrict__ Wbf,
                                                const int* __restrict__ idx,
                                                const float* __restrict__ p,
                                                int node, int coutTile, int lane) {
    const int e = lane & 15, h = lane >> 4;
    const int bbase = (node / NN) * NN;
    const int nb = bbase + idx[(size_t)node * KK + e];
    const __bf16* row = hbf + (size_t)nb * CPAD;
    v8f acc = {};
#pragma unroll
    for (int k0 = 0; k0 < CPAD; k0 += 32) {
        v16bf a;
#pragma unroll
        for (int i = 0; i < 8; ++i) a[i] = row[k0 + h * 8 + i];
#pragma unroll
        for (int i = 0; i < 8; ++i) a[8 + i] = row[k0 + 16 + h * 8 + i];
        v16bf bfrag = load_bT_bf16(Wbf, CPAD, coutTile * 16, k0, lane);
        acc = wmma_bf16(a, bfrag, acc);
    }
    float pv = p[(size_t)node * OUTC + coutTile * 16 + (lane & 15)];
#pragma unroll
    for (int v = 0; v < 8; ++v) acc[v] += pv;
    return acc;
}

// ---- pass 1: accumulate per-channel sum / sum-of-squares over all edges ----
template <int CPAD>
__global__ __launch_bounds__(256) void edge_stats_kernel(const __bf16* __restrict__ hbf,
                                                         const __bf16* __restrict__ Wbf,
                                                         const int* __restrict__ idx,
                                                         const float* __restrict__ p,
                                                         float* __restrict__ gsum, float* __restrict__ gsq, int Cout) {
    __shared__ float lsum[256], lsq2[256];
    lsum[threadIdx.x] = 0.f; lsq2[threadIdx.x] = 0.f;
    __syncthreads();
    const int lane = threadIdx.x & 31;
    const int wid = blockIdx.x * 8 + (threadIdx.x >> 5);
    const int ctiles = Cout >> 4;
    const int node = wid / ctiles, coutTile = wid % ctiles;
    v8f acc = edge_tile<CPAD>(hbf, Wbf, idx, p, node, coutTile, lane);
    float s = 0.f, s2 = 0.f;
#pragma unroll
    for (int v = 0; v < 8; ++v) { s += acc[v]; s2 += acc[v] * acc[v]; }
    s  += __shfl_xor(s, 16);
    s2 += __shfl_xor(s2, 16);
    if (lane < 16) {
        int o = coutTile * 16 + lane;
        atomicAdd(&lsum[o], s);
        atomicAdd(&lsq2[o], s2);
    }
    __syncthreads();
    if (threadIdx.x < Cout) {
        atomicAdd(&gsum[threadIdx.x], lsum[threadIdx.x]);
        atomicAdd(&gsq[threadIdx.x], lsq2[threadIdx.x]);
    }
}

__global__ void finalize_stats_kernel(const float* __restrict__ gsum, const float* __restrict__ gsq,
                                      const float* __restrict__ gamma, const float* __restrict__ beta,
                                      float* __restrict__ scale, float* __restrict__ shift, int Cout) {
    int o = threadIdx.x;
    if (o >= Cout) return;
    const float invCnt = 1.0f / (float)((size_t)BN * KK);
    float mean = gsum[o] * invCnt;
    float var  = gsq[o] * invCnt - mean * mean;
    float istd = rsqrtf(var + BN_EPS);
    float sc = gamma[o] * istd;
    scale[o] = sc;
    shift[o] = beta[o] - mean * sc;
}

// ---- pass 2: recompute tile, batchnorm, max over 16 edges, leaky-relu ----
template <int CPAD>
__global__ __launch_bounds__(256) void edge_max_kernel(const __bf16* __restrict__ hbf,
                                                       const __bf16* __restrict__ Wbf,
                                                       const int* __restrict__ idx,
                                                       const float* __restrict__ p,
                                                       const float* __restrict__ scale, const float* __restrict__ shift,
                                                       float* __restrict__ hnext, int Cout) {
    const int lane = threadIdx.x & 31;
    const int wid = blockIdx.x * 8 + (threadIdx.x >> 5);
    const int ctiles = Cout >> 4;
    const int node = wid / ctiles, coutTile = wid % ctiles;
    v8f acc = edge_tile<CPAD>(hbf, Wbf, idx, p, node, coutTile, lane);
    const int o = coutTile * 16 + (lane & 15);
    float sc = scale[o], sh = shift[o];
    float m = -1e30f;
#pragma unroll
    for (int v = 0; v < 8; ++v) m = fmaxf(m, acc[v] * sc + sh);
    m = fmaxf(m, __shfl_xor(m, 16));
    if (lane < 16) {
        float y = (m >= 0.f) ? m : 0.2f * m;
        hnext[(size_t)node * OUTC + o] = y;
    }
}

__global__ void zero_kernel(float* __restrict__ ptr, int n) {
    int i = blockIdx.x * 256 + threadIdx.x;
    if (i < n) ptr[i] = 0.f;
}

// ---- attention pooling: a = softmax_n(gate), out = sum_n a * feat ----
__global__ __launch_bounds__(256) void pool_kernel(const float* __restrict__ gate,
                                                   const float* __restrict__ feat,
                                                   float* __restrict__ out) {
    __shared__ float r1[256], r2[256];
    const int b = blockIdx.x >> 8, o = blockIdx.x & 255, tid = threadIdx.x;
    const float* g = gate + (size_t)b * NN * OUTC + o;
    const float* f = feat + (size_t)b * NN * OUTC + o;
    float mx = -1e30f;
    for (int n = tid; n < NN; n += 256) mx = fmaxf(mx, g[(size_t)n * OUTC]);
    r1[tid] = mx; __syncthreads();
    for (int s = 128; s > 0; s >>= 1) { if (tid < s) r1[tid] = fmaxf(r1[tid], r1[tid + s]); __syncthreads(); }
    float M = r1[0];
    __syncthreads();
    float se = 0.f, sw = 0.f;
    for (int n = tid; n < NN; n += 256) {
        float e = __expf(g[(size_t)n * OUTC] - M);
        se += e;
        sw += e * f[(size_t)n * OUTC];
    }
    r1[tid] = se; r2[tid] = sw; __syncthreads();
    for (int s = 128; s > 0; s >>= 1) {
        if (tid < s) { r1[tid] += r1[tid + s]; r2[tid] += r2[tid + s]; }
        __syncthreads();
    }
    if (tid == 0) out[(size_t)b * OUTC + o] = r2[0] / r1[0];
}

// ---------------- host-side dispatch helpers ----------------
static void launch_knn(int Cpad, const __bf16* hbf, const float* sq, int* idx, hipStream_t s) {
    dim3 g(NN / 16, BB);
    switch (Cpad) {
        case 32:  knn_kernel<32><<<g, 32, 0, s>>>(hbf, sq, idx); break;
        case 64:  knn_kernel<64><<<g, 32, 0, s>>>(hbf, sq, idx); break;
        case 128: knn_kernel<128><<<g, 32, 0, s>>>(hbf, sq, idx); break;
    }
}

static void launch_node_gemm(int Cpad, const __bf16* hbf, const __bf16* Wbf, const float* b1,
                             const float* b2, float* out, int Cout, int relu, hipStream_t s) {
    int blocks = ((BN / 16) * (Cout / 16)) / 8;
    switch (Cpad) {
        case 32:  node_gemm_kernel<32><<<blocks, 256, 0, s>>>(hbf, Wbf, b1, b2, out, Cout, relu); break;
        case 64:  node_gemm_kernel<64><<<blocks, 256, 0, s>>>(hbf, Wbf, b1, b2, out, Cout, relu); break;
        case 128: node_gemm_kernel<128><<<blocks, 256, 0, s>>>(hbf, Wbf, b1, b2, out, Cout, relu); break;
        case 256: node_gemm_kernel<256><<<blocks, 256, 0, s>>>(hbf, Wbf, b1, b2, out, Cout, relu); break;
    }
}

static void launch_edge_stats(int Cpad, const __bf16* hbf, const __bf16* Wbf, const int* idx,
                              const float* p, float* gsum, float* gsq, int Cout, hipStream_t s) {
    int blocks = (BN * (Cout / 16)) / 8;
    switch (Cpad) {
        case 32:  edge_stats_kernel<32><<<blocks, 256, 0, s>>>(hbf, Wbf, idx, p, gsum, gsq, Cout); break;
        case 64:  edge_stats_kernel<64><<<blocks, 256, 0, s>>>(hbf, Wbf, idx, p, gsum, gsq, Cout); break;
        case 128: edge_stats_kernel<128><<<blocks, 256, 0, s>>>(hbf, Wbf, idx, p, gsum, gsq, Cout); break;
    }
}

static void launch_edge_max(int Cpad, const __bf16* hbf, const __bf16* Wbf, const int* idx,
                            const float* p, const float* scale, const float* shift,
                            float* hnext, int Cout, hipStream_t s) {
    int blocks = (BN * (Cout / 16)) / 8;
    switch (Cpad) {
        case 32:  edge_max_kernel<32><<<blocks, 256, 0, s>>>(hbf, Wbf, idx, p, scale, shift, hnext, Cout); break;
        case 64:  edge_max_kernel<64><<<blocks, 256, 0, s>>>(hbf, Wbf, idx, p, scale, shift, hnext, Cout); break;
        case 128: edge_max_kernel<128><<<blocks, 256, 0, s>>>(hbf, Wbf, idx, p, scale, shift, hnext, Cout); break;
    }
}

extern "C" void kernel_launch(void* const* d_in, const int* in_sizes, int n_in,
                              void* d_out, int out_size, void* d_ws, size_t ws_size,
                              hipStream_t stream) {
    const float* x      = (const float*)d_in[0];
    const float* feat_W = (const float*)d_in[25];
    const float* feat_b = (const float*)d_in[26];
    const float* gat_W  = (const float*)d_in[27];
    const float* gat_b  = (const float*)d_in[28];
    const int emb[4] = {64, 64, 128, 256};

    char* ws = (char*)d_ws;
    float*  hcur = (float*)ws;   ws += (size_t)BN * OUTC * 4;
    float*  hnxt = (float*)ws;   ws += (size_t)BN * OUTC * 4;
    __bf16* hbf  = (__bf16*)ws;  ws += (size_t)BN * OUTC * 2;
    float*  pbuf = (float*)ws;   ws += (size_t)BN * OUTC * 4;   // node term, later: gate
    float*  fbuf = (float*)ws;   ws += (size_t)BN * OUTC * 4;   // feat
    float*  sq   = (float*)ws;   ws += (size_t)BN * 4;
    int*    idx  = (int*)ws;     ws += (size_t)BN * KK * 4;
    float*  stats = (float*)ws;  ws += 1024 * 4;                // [sum|sumsq|scale|shift] x 256
    __bf16* wtb  = (__bf16*)ws;  ws += (size_t)OUTC * OUTC * 2; // Wt (bf16, padded)
    __bf16* wdb  = (__bf16*)ws;                                  // Wp - Wt (bf16, padded)

    for (int L = 0; L < 4; ++L) {
        const int Cin  = (L == 0) ? 3  : emb[L - 1];
        const int Cpad = (L == 0) ? 32 : Cin;
        const int Cout = emb[L];
        const float* src = (L == 0) ? x : hcur;
        const int srcStride = (L == 0) ? 3 : OUTC;
        const float* Wt    = (const float*)d_in[1 + 6 * L + 0];
        const float* bt    = (const float*)d_in[1 + 6 * L + 1];
        const float* Wp    = (const float*)d_in[1 + 6 * L + 2];
        const float* bp    = (const float*)d_in[1 + 6 * L + 3];
        const float* gamma = (const float*)d_in[1 + 6 * L + 4];
        const float* beta  = (const float*)d_in[1 + 6 * L + 5];

        int wtot = Cout * Cpad;
        convw_kernel<<<(wtot + 255) / 256, 256, 0, stream>>>(Wt, nullptr, Cin, Cpad, wtot, wtb);
        convw_kernel<<<(wtot + 255) / 256, 256, 0, stream>>>(Wp, Wt, Cin, Cpad, wtot, wdb);
        prep_kernel<<<BN / 256, 256, 0, stream>>>(src, srcStride, Cin, Cpad, hbf, sq);
        launch_knn(Cpad, hbf, sq, idx, stream);

        // p = h*(Wp-Wt)^T + bt + bp
        launch_node_gemm(Cpad, hbf, wdb, bt, bp, pbuf, Cout, 0, stream);

        zero_kernel<<<2, 256, 0, stream>>>(stats, 512);
        launch_edge_stats(Cpad, hbf, wtb, idx, pbuf, stats, stats + 256, Cout, stream);
        finalize_stats_kernel<<<1, 256, 0, stream>>>(stats, stats + 256, gamma, beta,
                                                     stats + 512, stats + 768, Cout);
        launch_edge_max(Cpad, hbf, wtb, idx, pbuf, stats + 512, stats + 768, hnxt, Cout, stream);
        float* t = hcur; hcur = hnxt; hnxt = t;
    }

    prep_kernel<<<BN / 256, 256, 0, stream>>>(hcur, OUTC, OUTC, OUTC, hbf, sq);
    convw_kernel<<<(OUTC * OUTC + 255) / 256, 256, 0, stream>>>(gat_W, nullptr, OUTC, OUTC, OUTC * OUTC, wtb);
    convw_kernel<<<(OUTC * OUTC + 255) / 256, 256, 0, stream>>>(feat_W, nullptr, OUTC, OUTC, OUTC * OUTC, wdb);
    launch_node_gemm(256, hbf, wtb, gat_b, nullptr, pbuf, OUTC, 1, stream);
    launch_node_gemm(256, hbf, wdb, feat_b, nullptr, fbuf, OUTC, 1, stream);
    pool_kernel<<<BB * OUTC, 256, 0, stream>>>(pbuf, fbuf, (float*)d_out);
}
